// ModularNetController_26645977105099
// MI455X (gfx1250) — matrix operand order repaired and checked by reading, not compiled, not run on hardware
//
#include <hip/hip_runtime.h>
#include <hip/hip_bf16.h>

// Problem constants (from reference)
constexpr int NB   = 16;   // batch
constexpr int CIN  = 256;
constexpr int COUT = 256;
constexpr int HH   = 32;
constexpr int WW   = 32;
constexpr int NE   = 8;    // experts

typedef __attribute__((ext_vector_type(16))) __bf16 v16bf;
typedef __attribute__((ext_vector_type(8)))  float  v8f;

// Workspace layout (bytes)
constexpr size_t DEC_OFF   = 16384;                         // after pooled[4096] f32
constexpr size_t WBF_OFF   = 32768;
constexpr size_t WBF_ELEMS = (size_t)NE * 9 * COUT * CIN;   // 4,718,592 bf16 (~9 MB)
constexpr size_t XBF_OFF   = WBF_OFF + WBF_ELEMS * 2;       // 16B aligned
constexpr size_t XBF_ELEMS = (size_t)NB * HH * WW * CIN;    // 4,194,304 bf16 (~8 MB)

// ---------------------------------------------------------------------------
// CDNA5 async global->LDS copy helpers (ASYNCcnt-tracked; ISA 15.18.3 / 08_async)
// ---------------------------------------------------------------------------
__device__ __forceinline__ void async_copy_16k(const __bf16* __restrict__ g,
                                               void* lds_dst, int t) {
    // 16 KB = 1024 x 16B segments; 256 threads x 4 segments each.
    unsigned    l  = (unsigned)(size_t)lds_dst + (unsigned)t * 16u;
    const char* gp = (const char*)g + (size_t)t * 16;
    #pragma unroll
    for (int k = 0; k < 4; ++k) {
        asm volatile("global_load_async_to_lds_b128 %0, %1, off"
                     :: "v"(l), "v"(gp) : "memory");
        l  += 4096;
        gp += 4096;
    }
}

__device__ __forceinline__ void async_wait_all() {
    asm volatile("s_wait_asynccnt 0x0" ::: "memory");
}

__device__ __forceinline__ void zero_fill_16k(void* lds, int t) {
    int4 z; z.x = z.y = z.z = z.w = 0;
    int4* p = (int4*)lds;
    #pragma unroll
    for (int k = 0; k < 4; ++k) p[t + k * 256] = z;
}

// ---------------------------------------------------------------------------
// One-time layout conversions into workspace (bf16).
// wbf[(((e*9 + kh*3+kw)*8 + kc)*COUT + co)*32 + c]  <-  we[e][co][kc*32+c][kh][kw]
// ---------------------------------------------------------------------------
__global__ void __launch_bounds__(256)
convert_w_kernel(const float* __restrict__ we, __bf16* __restrict__ wbf) {
    const int idx = blockIdx.x * 256 + threadIdx.x;      // < 4,718,592
    const int c   = idx & 31;
    const int co  = (idx >> 5) & 255;
    const int rem = idx >> 13;                           // 0..575
    const int kc  = rem & 7;
    const int q   = rem >> 3;                            // e*9 + khw
    const int khw = q % 9;
    const int e   = q / 9;
    wbf[idx] = (__bf16)we[((size_t)(e * COUT + co) * CIN + kc * 32 + c) * 9 + khw];
}

// xbf[((i*HH + r)*WW + w)*CIN + ci]  <-  x[i][ci][r][w]   (LDS-tiled transpose)
__global__ void __launch_bounds__(256)
convert_x_kernel(const float* __restrict__ x, __bf16* __restrict__ xbf) {
    __shared__ __bf16 buf[WW][CIN + 2];
    const int r = blockIdx.x & 31;
    const int i = blockIdx.x >> 5;
    const int t = threadIdx.x;
    const float* xp = x + (size_t)i * CIN * HH * WW + (size_t)r * WW;
    for (int idx = t; idx < CIN * WW; idx += 256) {
        const int ci = idx >> 5, w = idx & 31;           // coalesced global read
        buf[w][ci] = (__bf16)xp[(size_t)ci * HH * WW + w];
    }
    __syncthreads();
    __bf16* op = xbf + (size_t)(i * HH + r) * WW * CIN;
    for (int idx = t; idx < CIN * WW; idx += 256)        // coalesced global write
        op[idx] = buf[idx >> 8][idx & 255];
}

// ---------------------------------------------------------------------------
// Controller: per-channel means, then logits + argmax.
// ---------------------------------------------------------------------------
__global__ void __launch_bounds__(256)
pool_kernel(const float* __restrict__ x, float* __restrict__ pooled) {
    __shared__ float sdata[256];
    const int ch = blockIdx.x;                           // 0..NB*CIN-1
    const float* p = x + (size_t)ch * (HH * WW);
    const int t = threadIdx.x;
    sdata[t] = p[t] + p[t + 256] + p[t + 512] + p[t + 768];
    __syncthreads();
    for (int off = 128; off > 0; off >>= 1) {
        if (t < off) sdata[t] += sdata[t + off];
        __syncthreads();
    }
    if (t == 0) pooled[ch] = sdata[0] * (1.0f / (HH * WW));
}

__global__ void __launch_bounds__(128)
decide_kernel(const float* __restrict__ pooled, const float* __restrict__ wc,
              const float* __restrict__ bc, int* __restrict__ dec) {
    __shared__ float lg[NB][NE];
    const int t = threadIdx.x;
    if (t < NB * NE) {
        const int b = t / NE, e = t % NE;
        float s = bc[e];
        for (int c = 0; c < CIN; ++c) s += pooled[b * CIN + c] * wc[e * CIN + c];
        lg[b][e] = s;
    }
    __syncthreads();
    if (t < NB) {
        int best = 0;
        float bv = lg[t][0];
        for (int e = 1; e < NE; ++e) {
            float v = lg[t][e];
            if (v > bv) { bv = v; best = e; }
        }
        dec[t] = best;
    }
}

// ---------------------------------------------------------------------------
// Implicit-GEMM 3x3 SAME conv via v_wmma_f32_16x16x32_bf16.
// Grid (h=32, e=8, i=16); 256 threads = 8 waves. Block computes all C_OUT=256
// for one output row of image i with expert e, stores to every j using e.
// ---------------------------------------------------------------------------
__global__ void __launch_bounds__(256)
moe_conv_kernel(const __bf16* __restrict__ xbf, const __bf16* __restrict__ wbf,
                const float* __restrict__ be, const int* __restrict__ dec,
                float* __restrict__ out) {
    const int h = blockIdx.x;
    const int e = blockIdx.y;
    const int i = blockIdx.z;

    unsigned jmask = 0u;
    for (int j = 0; j < NB; ++j) jmask |= (dec[j] == e) ? (1u << j) : 0u;
    if (jmask == 0u) return;                 // uniform early exit (no barriers yet)

    __shared__ alignas(16) __bf16 Xs[WW][CIN];         // 16 KB: [col][ci]
    __shared__ alignas(16) __bf16 Wsh[3][COUT][32];    // 48 KB: [kw][co][ci_off]

    const int t    = threadIdx.x;
    const int lane = t & 31;
    const int wid  = t >> 5;                 // wave -> co rows [wid*32, wid*32+32)
    const int n    = lane & 15;
    const bool hi  = lane >= 16;

    v8f acc[2][2];
    #pragma unroll
    for (int mt = 0; mt < 2; ++mt)
        #pragma unroll
        for (int nt = 0; nt < 2; ++nt)
            #pragma unroll
            for (int q = 0; q < 8; ++q) acc[mt][nt][q] = 0.0f;

    #pragma clang loop unroll(disable)
    for (int kh = 0; kh < 3; ++kh) {
        const int r = h + kh - 1;            // SAME padding source row

        __syncthreads();                     // prior reads of Xs/Wsh complete
        if (r >= 0 && r < HH)
            async_copy_16k(xbf + (size_t)(i * HH + r) * WW * CIN, Xs, t);
        else
            zero_fill_16k(Xs, t);

        #pragma clang loop unroll(disable)
        for (int kc = 0; kc < 8; ++kc) {
            if (kc > 0) __syncthreads();     // prior Wsh reads complete
            #pragma unroll
            for (int kw = 0; kw < 3; ++kw)
                async_copy_16k(wbf + (size_t)((e * 9 + kh * 3 + kw) * 8 + kc) * 8192,
                               Wsh[kw], t);
            async_wait_all();                // this wave's async copies landed
            __syncthreads();                 // all waves' copies landed

            #pragma unroll
            for (int kw = 0; kw < 3; ++kw) {
                // A fragments (weights 16x32): lane m = lane%16,
                // VGPR pair p: K = (p<4?0:16) + (hi?8:0) + 2*(p%4)
                v16bf a[2];
                #pragma unroll
                for (int mt = 0; mt < 2; ++mt) {
                    const __bf16* wrow = &Wsh[kw][wid * 32 + mt * 16 + n][0];
                    #pragma unroll
                    for (int p8 = 0; p8 < 8; ++p8) {
                        const int k0 = ((p8 < 4) ? 0 : 16) + (hi ? 8 : 0) + 2 * (p8 & 3);
                        a[mt][2 * p8]     = wrow[k0];
                        a[mt][2 * p8 + 1] = wrow[k0 + 1];
                    }
                }
                // B fragments (x 32x16): lane n = col; K = (q<8?0:16) + (hi?8:0) + q%8
                v16bf bfr[2];
                #pragma unroll
                for (int nt = 0; nt < 2; ++nt) {
                    const int col = nt * 16 + n + kw - 1;
                    if (col >= 0 && col < WW) {
                        const __bf16* xr = &Xs[col][kc * 32 + (hi ? 8 : 0)];
                        #pragma unroll
                        for (int q = 0; q < 16; ++q) bfr[nt][q] = xr[(q & 7) + ((q >> 3) << 4)];
                    } else {
                        #pragma unroll
                        for (int q = 0; q < 16; ++q) bfr[nt][q] = (__bf16)0.0f;
                    }
                }
                #pragma unroll
                for (int mt = 0; mt < 2; ++mt)
                    #pragma unroll
                    for (int nt = 0; nt < 2; ++nt)
                        acc[mt][nt] = __builtin_amdgcn_wmma_f32_16x16x32_bf16(
                            false, a[mt], false, bfr[nt],
                            (short)0, acc[mt][nt], false, false);
            }
        }
    }

    // Epilogue: bias + relu once into registers, then fan out stores per j.
    const float* beE = be + e * COUT;
    float res[2][2][8];
    #pragma unroll
    for (int mt = 0; mt < 2; ++mt)
        #pragma unroll
        for (int rr = 0; rr < 8; ++rr) {
            const int co = wid * 32 + mt * 16 + rr + (hi ? 8 : 0);
            const float bias = beE[co];
            res[mt][0][rr] = fmaxf(acc[mt][0][rr] + bias, 0.0f);
            res[mt][1][rr] = fmaxf(acc[mt][1][rr] + bias, 0.0f);
        }

    #pragma clang loop unroll(disable)
    for (int j = 0; j < NB; ++j) {
        if (!((jmask >> j) & 1u)) continue;
        float* op = out + (size_t)(j * NB + i) * (COUT * HH * WW) + (size_t)h * WW;
        #pragma unroll
        for (int mt = 0; mt < 2; ++mt)
            #pragma unroll
            for (int nt = 0; nt < 2; ++nt)
                #pragma unroll
                for (int rr = 0; rr < 8; ++rr) {
                    const int co = wid * 32 + mt * 16 + rr + (hi ? 8 : 0);
                    op[(size_t)co * (HH * WW) + nt * 16 + n] = res[mt][nt][rr];
                }
    }
}

// ---------------------------------------------------------------------------
extern "C" void kernel_launch(void* const* d_in, const int* in_sizes, int n_in,
                              void* d_out, int out_size, void* d_ws, size_t ws_size,
                              hipStream_t stream) {
    const float* x  = (const float*)d_in[0];
    const float* wc = (const float*)d_in[1];
    const float* bc = (const float*)d_in[2];
    const float* we = (const float*)d_in[3];
    const float* be = (const float*)d_in[4];
    float* out = (float*)d_out;

    float*  pooled = (float*)d_ws;
    int*    dec    = (int*)((char*)d_ws + DEC_OFF);
    __bf16* wbf    = (__bf16*)((char*)d_ws + WBF_OFF);
    __bf16* xbf    = (__bf16*)((char*)d_ws + XBF_OFF);

    convert_w_kernel<<<(int)(WBF_ELEMS / 256), 256, 0, stream>>>(we, wbf);
    convert_x_kernel<<<NB * HH, 256, 0, stream>>>(x, xbf);
    pool_kernel<<<NB * CIN, 256, 0, stream>>>(x, pooled);
    decide_kernel<<<1, 128, 0, stream>>>(pooled, wc, bc, dec);

    dim3 grid(HH, NE, NB);
    moe_conv_kernel<<<grid, 256, 0, stream>>>(xbf, wbf, be, dec, out);
}
